// GELU184_39857296507248
// MI455X (gfx1250) — compile-verified, matrix operand order checked by reading, not run on hardware
//
#include <hip/hip_runtime.h>
#include <hip/hip_bf16.h>
#include <math.h>

typedef __attribute__((ext_vector_type(16))) _Float16 v16h;
typedef __attribute__((ext_vector_type(8)))  _Float16 v8h;
typedef __attribute__((ext_vector_type(8)))  float    v8f;
typedef int v4i __attribute__((vector_size(16)));
typedef __attribute__((address_space(1))) v4i* gv4i_p;   // global int4*
typedef __attribute__((address_space(3))) v4i* lv4i_p;   // LDS int4*

#define B_ 8
#define T_ 2048
#define D_ 512
#define GELU_C 0.7978845608028654f

// CDNA5 async global->LDS path (ASYNCcnt) if the toolchain exposes it.
#if defined(__has_builtin)
#if __has_builtin(__builtin_amdgcn_global_load_async_to_lds_b128)
#define HAVE_ASYNC_LDS 1
#endif
#endif
#ifndef HAVE_ASYNC_LDS
#define HAVE_ASYNC_LDS 0
#endif

// Scheduler pinning (mask: 0x008 = MFMA/WMMA, 0x100 = DS read, 0x010 = VMEM).
#if defined(__has_builtin)
#if __has_builtin(__builtin_amdgcn_sched_group_barrier)
#define SGB(mask, size, id) __builtin_amdgcn_sched_group_barrier(mask, size, id)
#endif
#endif
#ifndef SGB
#define SGB(mask, size, id)
#endif

__device__ __forceinline__ void wait_asynccnt0() {
#if defined(__has_builtin)
#if __has_builtin(__builtin_amdgcn_s_wait_asynccnt)
    __builtin_amdgcn_s_wait_asynccnt(0);
    return;
#else
    asm volatile("s_wait_asynccnt 0x0" ::: "memory");
    return;
#endif
#else
    asm volatile("s_wait_asynccnt 0x0" ::: "memory");
#endif
}

__device__ __forceinline__ float gelu_f(float x) {
    float x3 = x * x * x;
    return 0.5f * x * (1.0f + tanhf(GELU_C * (x + 0.044715f * x3)));
}

// ---------------------------------------------------------------- reductions
__device__ __forceinline__ float block_sum_256(float v, float* smem) {
    #pragma unroll
    for (int m = 16; m >= 1; m >>= 1) v += __shfl_xor(v, m, 32);
    int wave = threadIdx.x >> 5;
    if ((threadIdx.x & 31) == 0) smem[wave] = v;
    __syncthreads();
    if (threadIdx.x == 0) {
        float t = 0.f;
        #pragma unroll
        for (int i = 0; i < 8; ++i) t += smem[i];
        smem[8] = t;
    }
    __syncthreads();
    float r = smem[8];
    __syncthreads();
    return r;
}

// ---------------------------------------------------------------- k0: ema_n
__global__ void k0_ema_norm(const float* __restrict__ ema,
                            float* __restrict__ ema_n) {
    __shared__ float s[512];
    int i = threadIdx.x;
    float e = ema[i];
    s[i] = e * e;
    __syncthreads();
    for (int st = 256; st > 0; st >>= 1) {
        if (i < st) s[i] += s[i + st];
        __syncthreads();
    }
    float inv = 1.0f / fmaxf(sqrtf(s[0]), 1e-12f);
    ema_n[i] = e * inv;
}

// --------------------------------------------- k1: x_n (f16) + gate_cos(row)
__global__ void k1_rowprep(const float* __restrict__ x,
                           const float* __restrict__ ema_n,
                           const float* __restrict__ log_tau,
                           _Float16* __restrict__ xn,
                           float* __restrict__ gate_cos) {
    __shared__ float smem[16];
    const int row = blockIdx.x;                 // b*T + t
    const float* xr = x + (size_t)row * D_;
    const int t0 = threadIdx.x * 2;             // 256 thr * 2 = 512

    float x0 = xr[t0], x1 = xr[t0 + 1];
    float sx = block_sum_256(x0 * x0 + x1 * x1, smem);
    float invx = 1.0f / fmaxf(sqrtf(sx), 1e-12f);

    _Float16* xnr = xn + (size_t)row * D_;
    xnr[t0]     = (_Float16)(x0 * invx);
    xnr[t0 + 1] = (_Float16)(x1 * invx);

    float o0 = gelu_f(x0), o1 = gelu_f(x1);
    float so = block_sum_256(o0 * o0 + o1 * o1, smem);
    float e0 = ema_n[t0], e1 = ema_n[t0 + 1];
    float sd = block_sum_256(o0 * e0 + o1 * e1, smem);

    if (threadIdx.x == 0) {
        float invo = 1.0f / fmaxf(sqrtf(so), 1e-12f);
        float cose = fminf(fmaxf(sd * invo, -1.0f), 1.0f);
        float tau  = expf(log_tau[0]);
        gate_cos[row] = expf(-tau * cose);
    }
}

// -------------------------------------- k2: WMMA similarity + per-row 1NN max
// 16-bit A-layout loader (B = transposed tile of the same row-major matrix,
// so the same layout applies with lane -> N):
// lane<16 : row = row0+lane, halves K = kk+[0..7] and kk+16+[0..7]
// lane>=16: row = row0+lane-16, halves K = kk+8+[0..7] and kk+24+[0..7]
__device__ __forceinline__ v16h gmem_frag(const _Float16* __restrict__ base,
                                          int row0, int kk, int lane) {
    int hi = (lane >> 4) & 1;
    int m  = lane & 15;
    const _Float16* p = base + (size_t)(row0 + m) * D_ + kk + (hi ? 8 : 0);
    v8h lo = *(const v8h*)(p);
    v8h hh = *(const v8h*)(p + 16);
    v16h r;
    #pragma unroll
    for (int i = 0; i < 8; ++i) { r[i] = lo[i]; r[8 + i] = hh[i]; }
    return r;
}

__device__ __forceinline__ v16h lds_frag(const _Float16* buf, int kk, int lane) {
    int hi = (lane >> 4) & 1;
    int m  = lane & 15;
    const _Float16* p = buf + m * D_ + kk + (hi ? 8 : 0);
    v8h lo = *(const v8h*)(p);
    v8h hh = *(const v8h*)(p + 16);
    v16h r;
    #pragma unroll
    for (int i = 0; i < 8; ++i) { r[i] = lo[i]; r[8 + i] = hh[i]; }
    return r;
}

__global__ void __launch_bounds__(128, 1)
k2_simmax(const _Float16* __restrict__ xn, float* __restrict__ nnmax) {
    // Block: 4 waves, 64 rows (one 16-row A tile per wave).
    // All waves sweep the same column tiles -> stage B tiles in LDS (double buf).
    __shared__ __align__(16) _Float16 lbuf[2][16 * D_];   // 2 x 16 KB

    const int tilesPerBatch = T_ / 16;                    // 128
    const int lane = threadIdx.x & 31;
    const int wv   = threadIdx.x >> 5;
    const int b    = blockIdx.x / (T_ / 64);
    const int rb   = (blockIdx.x % (T_ / 64)) * 64;       // block row base
    const int row0 = rb + wv * 16;                        // this wave's A rows
    const _Float16* base = xn + (size_t)b * T_ * D_;

    // Preload all 16 K-fragments of A (full K=512) -> 128 VGPRs, reused 128x.
    v16h af[16];
    #pragma unroll
    for (int k = 0; k < 16; ++k) af[k] = gmem_frag(base, row0, k * 32, lane);

    float rmax[8];
    #pragma unroll
    for (int v = 0; v < 8; ++v) rmax[v] = -3.0f;

    const int hi   = (lane >> 4) & 1;
    const int n15  = lane & 15;
    const int mrow = hi ? 8 : 0;

    // A column tile (16 rows x 512 halves) is 16 KB *contiguous* in memory:
    // each of the 128 threads moves 8 x 16B chunks (stride 2 KB = 128 int4).
#if HAVE_ASYNC_LDS
    auto async_tile = [&](int col0, int nb) {
        gv4i_p gsrc = (gv4i_p)(void*)(base + (size_t)col0 * D_);
        lv4i_p ldst = (lv4i_p)(void*)(&lbuf[nb][0]);
        gsrc += threadIdx.x;
        ldst += threadIdx.x;
        #pragma unroll
        for (int i = 0; i < 8; ++i)
            __builtin_amdgcn_global_load_async_to_lds_b128(
                gsrc + i * 128, ldst + i * 128, 0, 0);
    };
    async_tile(0, 0);
    wait_asynccnt0();
    __syncthreads();
#else
    uint4 g[8];
    auto load_tile = [&](int col0) {
        const uint4* src = (const uint4*)(base + (size_t)col0 * D_);
        #pragma unroll
        for (int i = 0; i < 8; ++i) g[i] = src[threadIdx.x + i * 128];
    };
    auto store_tile = [&](int nb) {
        uint4* dst = (uint4*)(&lbuf[nb][0]);
        #pragma unroll
        for (int i = 0; i < 8; ++i) dst[threadIdx.x + i * 128] = g[i];
    };
    load_tile(0);
    store_tile(0);
    __syncthreads();
#endif

    for (int jt = 0; jt < tilesPerBatch; ++jt) {
        const int cur  = jt & 1;
        const int col0 = jt * 16;

        // Kick off the NEXT tile's copy (unconditional, wrapping, so the whole
        // tile body stays a single scheduling region for the recipe below).
#if HAVE_ASYNC_LDS
        async_tile(((jt + 1) & (tilesPerBatch - 1)) * 16, 1 - cur);
#else
        if (jt + 1 < tilesPerBatch) load_tile((jt + 1) * 16);
#endif

        // Compute this tile from LDS with a 2-deep fragment pipeline over K.
        // Single accumulator chain: D->C accumulation is not a WMMA hazard.
        const _Float16* buf = &lbuf[cur][0];
        v8f c = {};
        v16h bf0 = lds_frag(buf, 0, lane);
        #pragma unroll
        for (int k = 0; k < 16; ++k) {
            v16h bf1;
            if (k + 1 < 16) bf1 = lds_frag(buf, (k + 1) * 32, lane);
            c = __builtin_amdgcn_wmma_f32_16x16x32_f16(
                    false, af[k], false, bf0, (short)0, c, false, false);
            bf0 = bf1;
        }

        // Scheduling recipe for this region: async VMEM first, DS reads one
        // K-step ahead of their WMMA so waits become partial (dscnt<=2),
        // not full (dscnt==0).
        SGB(0x010, 8, 0);              // 8 async global->LDS (next tile)
        SGB(0x100, 4, 0);              // ds reads for k=0,1
        #pragma unroll
        for (int k = 0; k < 14; ++k) {
            SGB(0x008, 1, 0);          // wmma k
            SGB(0x100, 2, 0);          // ds reads for k+2
        }
        SGB(0x008, 1, 0);              // wmma 14
        SGB(0x008, 1, 0);              // wmma 15

        // Branchless diagonal mask (M==N on the diagonal tile).
        const bool diag_tile = (col0 == row0);
        #pragma unroll
        for (int v = 0; v < 8; ++v)
            c[v] = (diag_tile && ((v + mrow) == n15)) ? -2.0f : c[v];

        #pragma unroll
        for (int v = 0; v < 8; ++v) rmax[v] = fmaxf(rmax[v], c[v]);

#if HAVE_ASYNC_LDS
        wait_asynccnt0();              // next buffer fully landed in LDS
#else
        if (jt + 1 < tilesPerBatch) store_tile(1 - cur);
#endif
        __syncthreads();
    }

    // max across the 16 lanes of each wave-half (rows M=v+mrow)
    #pragma unroll
    for (int m = 8; m >= 1; m >>= 1) {
        #pragma unroll
        for (int v = 0; v < 8; ++v)
            rmax[v] = fmaxf(rmax[v], __shfl_xor(rmax[v], m, 32));
    }
    if (n15 == 0) {                    // lane 0 -> rows 0..7, lane 16 -> 8..15
        float* dst = nnmax + (size_t)b * T_ + row0 + mrow;
        #pragma unroll
        for (int v = 0; v < 8; ++v) dst[v] = rmax[v];
    }
}

// ------------------------------------------------- k3: fused gate * gelu(x)
__global__ void k3_apply(const float* __restrict__ x,
                         const float* __restrict__ gate_cos,
                         const float* __restrict__ nnmax,
                         const float* __restrict__ log_sigma_raw,
                         const float* __restrict__ log_w_raw,
                         float* __restrict__ out) {
    size_t i = (size_t)blockIdx.x * blockDim.x + threadIdx.x;
    size_t row = i / D_;
    float sigma = logf(1.0f + expf(log_sigma_raw[0]));   // softplus
    float w     = logf(1.0f + expf(log_w_raw[0]));
    float ns    = fminf(fmaxf(nnmax[row], -1.0f), 1.0f);
    float novelty  = (1.0f - ns) * 0.5f;
    float surprise = tanhf(sigma * novelty);
    float gate  = gate_cos[row] * (1.0f + w * surprise);
    out[i] = gelu_f(x[i]) * gate;
}

// ---------------------------------------------------------------- launcher
extern "C" void kernel_launch(void* const* d_in, const int* in_sizes, int n_in,
                              void* d_out, int out_size, void* d_ws, size_t ws_size,
                              hipStream_t stream) {
    (void)in_sizes; (void)n_in; (void)out_size; (void)ws_size;
    const float* x             = (const float*)d_in[0];
    // d_in[1] = logit_decay (unused in steady-state forward)
    const float* log_tau       = (const float*)d_in[2];
    const float* log_sigma_raw = (const float*)d_in[3];
    const float* log_w_raw     = (const float*)d_in[4];
    const float* ema_out       = (const float*)d_in[5];
    float* out = (float*)d_out;

    char* ws = (char*)d_ws;
    const size_t OFF_XN  = 0;                                      // 16 MB f16
    const size_t OFF_EMA = (size_t)B_ * T_ * D_ * sizeof(_Float16);
    const size_t OFF_GC  = OFF_EMA + 4096;                         // 64 KB
    const size_t OFF_NN  = OFF_GC + (size_t)B_ * T_ * sizeof(float);
    _Float16* xn      = (_Float16*)(ws + OFF_XN);
    float*    ema_n   = (float*)(ws + OFF_EMA);
    float*    gatecos = (float*)(ws + OFF_GC);
    float*    nnmax   = (float*)(ws + OFF_NN);

    k0_ema_norm<<<1, 512, 0, stream>>>(ema_out, ema_n);
    k1_rowprep<<<B_ * T_, 256, 0, stream>>>(x, ema_n, log_tau, xn, gatecos);
    k2_simmax<<<B_ * (T_ / 64), 128, 0, stream>>>(xn, nnmax);
    const size_t n = (size_t)B_ * T_ * D_;
    k3_apply<<<(unsigned)(n / 256), 256, 0, stream>>>(x, gatecos, nnmax,
                                                      log_sigma_raw, log_w_raw, out);
}
